// TriangleOutgoing_84430467105271
// MI455X (gfx1250) — matrix operand order repaired and checked by reading, not compiled, not run on hardware
//
#include <hip/hip_runtime.h>

typedef __attribute__((ext_vector_type(16))) __bf16          v16bf;
typedef __attribute__((ext_vector_type(8)))  float           v8f;
typedef __attribute__((ext_vector_type(16))) unsigned short  u16x16;
typedef __attribute__((ext_vector_type(8)))  unsigned short  u16x8;

#define HW   (384 * 384)        // 147456 spatial positions
#define CZ   128                // channels
#define NTOT ((long)CZ * HW)    // 18,874,368 elements
#define NP   1152               // partial-reduction blocks
#define WMAT 16384              // elements per 128x128 weight matrix
#define LNEPS 1e-5f

#define WMMA_BF16(A, B, C) \
  __builtin_amdgcn_wmma_f32_16x16x32_bf16(false, (A), false, (B), (short)0, (C), false, false)

__device__ __forceinline__ unsigned short f2bf(float f) {
  unsigned u = __builtin_bit_cast(unsigned, f);
  u += 0x7FFFu + ((u >> 16) & 1u);          // round-to-nearest-even
  return (unsigned short)(u >> 16);
}

__device__ __forceinline__ float sigmoidf(float x) {
  return 1.0f / (1.0f + __expf(-x));
}

// A-fragment (16x32, MxK) from bf16 row-major rows, k contiguous, stride ld.
// lane half h: element e -> K = k0 + 16h + e
__device__ __forceinline__ v16bf fragA_bf16(const unsigned short* __restrict__ base,
                                            long row, int ld, int k0, int lane) {
  int kb = ((lane >> 4) << 4);
  const unsigned short* p = base + row * ld + k0 + kb;
  u16x8 lo = *(const u16x8*)p;
  u16x8 hi = *(const u16x8*)(p + 8);
  u16x16 v;
#pragma unroll
  for (int e = 0; e < 8; ++e) { v[e] = lo[e]; v[e + 8] = hi[e]; }
  return __builtin_bit_cast(v16bf, v);
}

// B-fragment (32x16, KxN) from bf16 row-major rows b[n][k] (contraction over
// contiguous k). lane half h: element e -> K = k0 + 16h + e.
__device__ __forceinline__ v16bf fragB_bf16(const unsigned short* __restrict__ base,
                                            long row, int ld, int k0, int lane) {
  const unsigned short* p = base + row * ld + k0 + ((lane >> 4) << 4);
  u16x8 lo = *(const u16x8*)p;
  u16x8 hi = *(const u16x8*)(p + 8);
  u16x16 v;
#pragma unroll
  for (int e = 0; e < 8; ++e) { v[e] = lo[e]; v[e + 8] = hi[e]; }
  return __builtin_bit_cast(v16bf, v);
}

// B-fragment from f32 channel-major src[c*HW + p] with on-the-fly LayerNorm:
// z = (x - mu) * rs * w + b.  K axis = channel c, N axis = position p.
__device__ __forceinline__ v16bf fragB_norm(const float* __restrict__ src,
                                            const float* __restrict__ w,
                                            const float* __restrict__ b,
                                            float mu, float rs,
                                            int k0, long p, int lane) {
  int kb = k0 + ((lane >> 4) << 4);
  u16x16 v;
#pragma unroll
  for (int e = 0; e < 16; ++e) {
    long idx = (long)(kb + e) * HW + p;
    v[e] = f2bf((src[idx] - mu) * rs * w[idx] + b[idx]);
  }
  return __builtin_bit_cast(v16bf, v);
}

// ---------------- weight f32 -> bf16 one-time conversion --------------------

__global__ __launch_bounds__(256) void k_wcvt(const float* __restrict__ src,
                                              unsigned short* __restrict__ dst) {
  int t = blockIdx.x * 256 + threadIdx.x;    // 16384 threads per matrix
  dst[t] = f2bf(src[t]);
}

// ---------------- LayerNorm statistics (two-stage f32 reduction) -------------

__global__ __launch_bounds__(256) void k_stats_partial(const float* __restrict__ x,
                                                       float* __restrict__ part) {
  __shared__ float s1[256], s2[256];
  long tid    = (long)blockIdx.x * 256 + threadIdx.x;
  long stride = (long)gridDim.x * 256;
  float s = 0.f, q = 0.f;
  for (long i = tid; i < NTOT; i += stride) { float v = x[i]; s += v; q += v * v; }
  s1[threadIdx.x] = s; s2[threadIdx.x] = q;
  __syncthreads();
  for (int o = 128; o > 0; o >>= 1) {
    if (threadIdx.x < o) {
      s1[threadIdx.x] += s1[threadIdx.x + o];
      s2[threadIdx.x] += s2[threadIdx.x + o];
    }
    __syncthreads();
  }
  if (threadIdx.x == 0) { part[blockIdx.x] = s1[0]; part[NP + blockIdx.x] = s2[0]; }
}

__global__ __launch_bounds__(256) void k_stats_final(const float* __restrict__ part,
                                                     float* __restrict__ stats) {
  __shared__ float s1[256], s2[256];
  float s = 0.f, q = 0.f;
  for (int i = threadIdx.x; i < NP; i += 256) { s += part[i]; q += part[NP + i]; }
  s1[threadIdx.x] = s; s2[threadIdx.x] = q;
  __syncthreads();
  for (int o = 128; o > 0; o >>= 1) {
    if (threadIdx.x < o) {
      s1[threadIdx.x] += s1[threadIdx.x + o];
      s2[threadIdx.x] += s2[threadIdx.x + o];
    }
    __syncthreads();
  }
  if (threadIdx.x == 0) {
    float mu  = s1[0] / (float)NTOT;
    float var = s2[0] / (float)NTOT - mu * mu;
    stats[0] = mu;
    stats[1] = rsqrtf(var + LNEPS);
  }
}

// --------- Fused 5-way projection: a = sig(za1)*za2, b = sig(zb1)*zb2, g ----
// One wave owns 16 positions; Z fragments (K=128) built once, reused by all
// 5 weight matrices x 8 M-tiles -> 160 v_wmma per wave. Weights pre-converted
// to bf16 (wbf: matrices 0..4 = wa1,wa2,wb1,wb2,wg).

__global__ __launch_bounds__(256) void k_proj(
    const float* __restrict__ x, const float* __restrict__ ln1w, const float* __restrict__ ln1b,
    const unsigned short* __restrict__ wbf,
    const float* __restrict__ ba1, const float* __restrict__ ba2,
    const float* __restrict__ bb1, const float* __restrict__ bb2,
    const float* __restrict__ bg,  const float* __restrict__ stats,
    unsigned short* __restrict__ abf, unsigned short* __restrict__ bbf,
    float* __restrict__ gbuf) {
  int  lane = threadIdx.x & 31;
  int  wave = threadIdx.x >> 5;
  long p0   = ((long)blockIdx.x * 8 + wave) * 16;
  float mu = stats[0], rs = stats[1];
  long p    = p0 + (lane & 15);
  int  half = lane >> 4;
  int  n    = lane & 15;
  long wrow = lane & 15;   // A-fragment lane row within tile

  v16bf zf[4];
#pragma unroll
  for (int kk = 0; kk < 4; ++kk)
    zf[kk] = fragB_norm(x, ln1w, ln1b, mu, rs, kk * 32, p, lane);

  for (int mt = 0; mt < 8; ++mt) {
    int o0 = mt * 16;
    v8f a1{}, a2{}, b1{}, b2{}, gg{};
#pragma unroll
    for (int r = 0; r < 8; ++r) {
      int o = o0 + r + 8 * half;
      a1[r] = ba1[o]; a2[r] = ba2[o]; b1[r] = bb1[o]; b2[r] = bb2[o]; gg[r] = bg[o];
    }
#pragma unroll
    for (int kk = 0; kk < 4; ++kk) {
      int k0 = kk * 32;
      a1 = WMMA_BF16(fragA_bf16(wbf + 0 * WMAT, o0 + wrow, CZ, k0, lane), zf[kk], a1);
      a2 = WMMA_BF16(fragA_bf16(wbf + 1 * WMAT, o0 + wrow, CZ, k0, lane), zf[kk], a2);
      b1 = WMMA_BF16(fragA_bf16(wbf + 2 * WMAT, o0 + wrow, CZ, k0, lane), zf[kk], b1);
      b2 = WMMA_BF16(fragA_bf16(wbf + 3 * WMAT, o0 + wrow, CZ, k0, lane), zf[kk], b2);
      gg = WMMA_BF16(fragA_bf16(wbf + 4 * WMAT, o0 + wrow, CZ, k0, lane), zf[kk], gg);
    }
#pragma unroll
    for (int r = 0; r < 8; ++r) {
      int  o   = o0 + r + 8 * half;
      long idx = (long)o * HW + p0 + n;
      abf[idx]  = f2bf(sigmoidf(a1[r]) * a2[r]);
      bbf[idx]  = f2bf(sigmoidf(b1[r]) * b2[r]);
      gbuf[idx] = sigmoidf(gg[r]);
    }
  }
}

// --------- Triangle update: tri[c] = a[c] (384xK) . b[c]^T  (K=384) ---------
// Block = 4 waves covering a 128(i) x 128(j) tile; each wave owns 64x64
// (4x4 accumulators, 4-way A/B fragment reuse -> 16 WMMA per 16 b128 loads),
// 12 k-steps.

__global__ __launch_bounds__(128) void k_tri(const unsigned short* __restrict__ abf,
                                             const unsigned short* __restrict__ bbf,
                                             float* __restrict__ tri) {
  int  lane = threadIdx.x & 31;
  int  wave = threadIdx.x >> 5;
  int  c    = blockIdx.z;
  int  i0   = blockIdx.y * 128 + (wave >> 1) * 64;
  int  j0   = blockIdx.x * 128 + (wave & 1) * 64;
  const unsigned short* ac = abf + (long)c * HW;   // rows [i][k], ld=384
  const unsigned short* bc = bbf + (long)c * HW;   // rows [j][k], ld=384
  int half = lane >> 4;
  int m    = lane & 15;

  v8f acc[4][4];
#pragma unroll
  for (int si = 0; si < 4; ++si)
#pragma unroll
    for (int sj = 0; sj < 4; ++sj) acc[si][sj] = v8f{};

#pragma unroll 2
  for (int k = 0; k < 12; ++k) {
    int k0 = k * 32;
    if (k < 10) {  // uniform branch: prefetch 2 k-steps ahead
      __builtin_prefetch(ac + ((long)i0 + m) * 384 + k0 + 64, 0, 1);
      __builtin_prefetch(bc + ((long)j0 + m) * 384 + k0 + 64, 0, 1);
    }
    v16bf A[4], B[4];
#pragma unroll
    for (int si = 0; si < 4; ++si)
      A[si] = fragA_bf16(ac, (long)i0 + si * 16 + m, 384, k0, lane);
#pragma unroll
    for (int sj = 0; sj < 4; ++sj)
      B[sj] = fragB_bf16(bc, (long)j0 + sj * 16 + m, 384, k0, lane);
#pragma unroll
    for (int si = 0; si < 4; ++si)
#pragma unroll
      for (int sj = 0; sj < 4; ++sj)
        acc[si][sj] = WMMA_BF16(A[si], B[sj], acc[si][sj]);
  }

  float* tc = tri + (long)c * HW;
#pragma unroll
  for (int si = 0; si < 4; ++si)
#pragma unroll
    for (int sj = 0; sj < 4; ++sj)
#pragma unroll
      for (int r = 0; r < 8; ++r)
        tc[(long)(i0 + si * 16 + r + 8 * half) * 384 + j0 + sj * 16 + m] = acc[si][sj][r];
}

// --------- Final: out = g * (wrep . LN2(tri) + brep) -----------------------

__global__ __launch_bounds__(256) void k_final(
    const float* __restrict__ tri, const float* __restrict__ ln2w, const float* __restrict__ ln2b,
    const unsigned short* __restrict__ wbf,   // bf16 wrep
    const float* __restrict__ brep, const float* __restrict__ g,
    const float* __restrict__ stats, float* __restrict__ out) {
  int  lane = threadIdx.x & 31;
  int  wave = threadIdx.x >> 5;
  long p0   = ((long)blockIdx.x * 8 + wave) * 16;
  float mu = stats[0], rs = stats[1];
  long p    = p0 + (lane & 15);
  int  half = lane >> 4;
  int  n    = lane & 15;
  long wrow = lane & 15;

  v16bf zf[4];
#pragma unroll
  for (int kk = 0; kk < 4; ++kk)
    zf[kk] = fragB_norm(tri, ln2w, ln2b, mu, rs, kk * 32, p, lane);

  for (int mt = 0; mt < 8; ++mt) {
    int o0 = mt * 16;
    v8f acc{};
#pragma unroll
    for (int r = 0; r < 8; ++r) acc[r] = brep[o0 + r + 8 * half];
#pragma unroll
    for (int kk = 0; kk < 4; ++kk)
      acc = WMMA_BF16(fragA_bf16(wbf, o0 + wrow, CZ, kk * 32, lane), zf[kk], acc);
#pragma unroll
    for (int r = 0; r < 8; ++r) {
      int  o   = o0 + r + 8 * half;
      long idx = (long)o * HW + p0 + n;
      out[idx] = g[idx] * acc[r];
    }
  }
}

// ---------------------------------------------------------------------------

extern "C" void kernel_launch(void* const* d_in, const int* in_sizes, int n_in,
                              void* d_out, int out_size, void* d_ws, size_t ws_size,
                              hipStream_t stream) {
  const float* x    = (const float*)d_in[0];
  const float* ln1w = (const float*)d_in[1];
  const float* ln1b = (const float*)d_in[2];
  const float* wa1  = (const float*)d_in[3];
  const float* ba1  = (const float*)d_in[4];
  const float* wa2  = (const float*)d_in[5];
  const float* ba2  = (const float*)d_in[6];
  const float* wb1  = (const float*)d_in[7];
  const float* bb1  = (const float*)d_in[8];
  const float* wb2  = (const float*)d_in[9];
  const float* bb2  = (const float*)d_in[10];
  const float* wg   = (const float*)d_in[11];
  const float* bg   = (const float*)d_in[12];
  const float* ln2w = (const float*)d_in[13];
  const float* ln2b = (const float*)d_in[14];
  const float* wrep = (const float*)d_in[15];
  const float* brep = (const float*)d_in[16];

  char* ws = (char*)d_ws;
  float*          statsX = (float*)ws;                 // 2 floats
  float*          statsT = (float*)(ws + 16);          // 2 floats
  float*          part   = (float*)(ws + 64);          // NP*2 floats
  unsigned short* wbf    = (unsigned short*)(ws + 16384);   // 6 x 32KB bf16 weights
  unsigned short* abf    = (unsigned short*)(ws + 262144);  // 37.75 MB
  unsigned short* bbf    = abf + NTOT;                      // 37.75 MB
  float*          gbuf   = (float*)(bbf + NTOT);            // 75.5 MB
  float*          tri    = gbuf + NTOT;                     // 75.5 MB
  float*          outf   = (float*)d_out;

  // 0) one-time bf16 conversion of the six 128x128 weight matrices
  k_wcvt<<<WMAT / 256, 256, 0, stream>>>(wa1,  wbf + 0 * WMAT);
  k_wcvt<<<WMAT / 256, 256, 0, stream>>>(wa2,  wbf + 1 * WMAT);
  k_wcvt<<<WMAT / 256, 256, 0, stream>>>(wb1,  wbf + 2 * WMAT);
  k_wcvt<<<WMAT / 256, 256, 0, stream>>>(wb2,  wbf + 3 * WMAT);
  k_wcvt<<<WMAT / 256, 256, 0, stream>>>(wg,   wbf + 4 * WMAT);
  k_wcvt<<<WMAT / 256, 256, 0, stream>>>(wrep, wbf + 5 * WMAT);

  // 1) LayerNorm-1 stats
  k_stats_partial<<<NP, 256, 0, stream>>>(x, part);
  k_stats_final<<<1, 256, 0, stream>>>(part, statsX);

  // 2) fused 5-way projection (normalize + 5 GEMMs + gating)
  k_proj<<<HW / 16 / 8, 256, 0, stream>>>(x, ln1w, ln1b, wbf,
                                          ba1, ba2, bb1, bb2, bg,
                                          statsX, abf, bbf, gbuf);

  // 3) 128 independent 384^3 GEMMs, 64x64 tile per wave
  k_tri<<<dim3(3, 3, 128), 128, 0, stream>>>(abf, bbf, tri);

  // 4) LayerNorm-2 stats
  k_stats_partial<<<NP, 256, 0, stream>>>(tri, part);
  k_stats_final<<<1, 256, 0, stream>>>(part, statsT);

  // 5) final projection + gate
  k_final<<<HW / 16 / 8, 256, 0, stream>>>(tri, ln2w, ln2b, wbf + 5 * WMAT,
                                           brep, gbuf, statsT, outf);
}
